// FakeQuantLinear_66726611911249
// MI455X (gfx1250) — matrix operand order repaired and checked by reading, not compiled, not run on hardware
//
#include <hip/hip_runtime.h>
#include <cstdint>

typedef __attribute__((ext_vector_type(16))) _Float16 v16h;
typedef __attribute__((ext_vector_type(8)))  _Float16 v8h;
typedef __attribute__((ext_vector_type(8)))  float    v8f;
typedef __attribute__((ext_vector_type(4)))  float    v4f;

#define TM   128   // M rows per workgroup tile
#define TN   128   // N cols per workgroup tile
#define KC   64    // K per LDS stage (2 MXFP blocks of 32)
#define LSTR 72    // halves per LDS row (64 + 8 pad for bank spread)

// ---- staged global load: 64 floats into registers ----
__device__ __forceinline__ void load_chunk(const float* __restrict__ g, v4f* d) {
#pragma unroll
  for (int i = 0; i < 16; ++i) d[i] = *(const v4f*)(g + 4 * i);
}

// ---- MXFP fake-quant of one 32-float block (from regs) -> exact f16 in LDS ----
__device__ __forceinline__ void quant_store_block(const v4f* d,
                                                  _Float16* __restrict__ l) {
  float amax = 0.0f;
#pragma unroll
  for (int i = 0; i < 8; ++i)
#pragma unroll
    for (int j = 0; j < 4; ++j) amax = fmaxf(amax, fabsf(d[i][j]));
  // shared_exp = floor(log2(amax)) - EMAX, clamp >= -127 (amax==0 -> -127)
  int eb = (int)((__float_as_uint(amax) >> 23) & 0xffu);
  int e  = (eb == 0) ? -127 : (eb - 135);
  if (e < -127) e = -127;
  const float scale = __builtin_ldexpf(1.0f, e);
  const float inv   = __builtin_ldexpf(1.0f, -e);
#pragma unroll
  for (int i = 0; i < 4; ++i) {
    v8h o;
#pragma unroll
    for (int j = 0; j < 8; ++j) {
      const int idx = i * 8 + j;
      float v  = d[idx >> 2][idx & 3] * inv;
      int   pe = (int)((__float_as_uint(v) >> 23) & 0xffu) - 127;  // floor(log2|v|)
      pe = pe < -6 ? -6 : pe;                                      // MIN_PEXP
      float q  = roundf(__builtin_ldexpf(v, 6 - pe));              // half-away, m=64
      float qv = __builtin_ldexpf(q, pe - 6);
      qv = fminf(fmaxf(qv, -448.0f), 448.0f);                      // MAX_NORM clip
      o[j] = (_Float16)(qv * scale);                               // exact rescale
    }
    *(v8h*)(l + 8 * i) = o;
  }
}

// ---- fragment loader (16B-aligned LDS, ISA 16-bit A/B layouts) ----
__device__ __forceinline__ v16h frag_gap(const _Float16* p, int gap) {
  v8h lo = *(const v8h*)p;
  v8h hi = *(const v8h*)(p + gap);
  return __builtin_shufflevector(lo, hi, 0, 1, 2, 3, 4, 5, 6, 7,
                                 8, 9, 10, 11, 12, 13, 14, 15);
}

__global__ __launch_bounds__(256) void fq_linear_kernel(
    const float* __restrict__ X, const float* __restrict__ W,
    const float* __restrict__ Bv, float* __restrict__ Out,
    int M, int N, int K) {
  // double-buffered stage: [buf][ x tile (TM rows) | w tile (TN rows) ]
  __shared__ alignas(32) _Float16 lds[2][(TM + TN) * LSTR];

  const int t    = threadIdx.x;
  const int wave = t >> 5;
  const int lane = t & 31;
  const int l15  = lane & 15;
  const int lhf  = lane >> 4;  // 0: lanes 0-15, 1: lanes 16-31

  const int rowBase = blockIdx.y * TM;
  const int colBase = blockIdx.x * TN;

  // this thread's quant source row (t<TM: x rows, else weight rows)
  const float* grow;
  int ldst;  // destination row offset within a buffer
  if (t < TM) {
    grow = X + (size_t)(rowBase + t) * K;
    ldst = t * LSTR;
  } else {
    grow = W + (size_t)(colBase + (t - TM)) * K;
    ldst = (TM + (t - TM)) * LSTR;
  }

  v8f acc[8] = {};
  v4f st[16];  // staged 64 floats for next chunk

  const int nk = K / KC;

  // prologue: stage + quant chunk 0 into buffer 0
  load_chunk(grow, st);
  quant_store_block(st,     &lds[0][ldst]);
  quant_store_block(st + 8, &lds[0][ldst + 32]);
  __syncthreads();

#pragma unroll 1
  for (int kc = 0; kc < nk; ++kc) {
    const int  cur  = kc & 1;
    const bool more = (kc + 1) < nk;

    // stage next chunk's global data early (latency hidden behind WMMAs)
    if (more) load_chunk(grow + (size_t)(kc + 1) * KC, st);
    if (kc + 2 < nk)
      __builtin_prefetch(grow + (size_t)(kc + 2) * KC, 0, 3);  // near-scope

    // ---- WMMA phase on buffer `cur` ----
    const _Float16* xb = &lds[cur][(wave * 16 + l15) * LSTR + lhf * 8];
    const _Float16* wb = &lds[cur][(TM + l15) * LSTR + lhf * 8];
    // A 16x32: lanes 0-15 K{0-7,16-23}, lanes 16-31 K{8-15,24-31}
    v16h a0 = frag_gap(xb, 16);       // K-slice 0 (K 0-31)
    v16h a1 = frag_gap(xb + 32, 16);  // K-slice 1 (K 32-63)
#pragma unroll
    for (int s = 0; s < 8; ++s) {
      // B 32x16: lane n = lane%16; lanes 0-15 K{0-7,16-23}, 16-31 K{8-15,24-31}
      const _Float16* wr = wb + s * 16 * LSTR;
      v16h b0 = frag_gap(wr, 16);
      v16h b1 = frag_gap(wr + 32, 16);
      // two back-to-back WMMAs per dscnt wait; D->C chaining has no hazard
      acc[s] = __builtin_amdgcn_wmma_f32_16x16x32_f16(
          false, a0, false, b0, (short)0, acc[s], false, false);
      acc[s] = __builtin_amdgcn_wmma_f32_16x16x32_f16(
          false, a1, false, b1, (short)0, acc[s], false, false);
    }

    // ---- quant next chunk into the other buffer (overlaps XDL drain) ----
    if (more) {
      _Float16* dst = &lds[1 - cur][ldst];
      quant_store_block(st,     dst);
      quant_store_block(st + 8, dst + 32);
    }
    __syncthreads();
  }

  // ---- epilogue: C/D layout (m = r + 8*lhf, n = lane%16) + bias ----
#pragma unroll
  for (int s = 0; s < 8; ++s) {
    const int   col = colBase + s * 16 + l15;
    const float bv  = Bv[col];
    const int   mr  = rowBase + wave * 16 + lhf * 8;
#pragma unroll
    for (int r = 0; r < 8; ++r) {
      Out[(size_t)(mr + r) * N + col] = acc[s][r] + bv;
    }
  }
}

extern "C" void kernel_launch(void* const* d_in, const int* in_sizes, int n_in,
                              void* d_out, int out_size, void* d_ws, size_t ws_size,
                              hipStream_t stream) {
  (void)n_in; (void)d_ws; (void)ws_size; (void)out_size;
  const float* x  = (const float*)d_in[0];
  const float* w  = (const float*)d_in[1];
  const float* bs = (const float*)d_in[2];
  float* out = (float*)d_out;

  const int N = in_sizes[2];                        // DOUT
  const int K = in_sizes[1] / N;                    // DIN
  const int M = (int)((long long)in_sizes[0] / K);  // B*S

  dim3 grid((unsigned)(N / TN), (unsigned)(M / TM));
  dim3 block(256);
  hipLaunchKernelGGL(fq_linear_kernel, grid, block, 0, stream,
                     x, w, bs, out, M, N, K);
}
